// VladPooling_20109036880611
// MI455X (gfx1250) — matrix operand order repaired and checked by reading, not compiled, not run on hardware
//
#include <hip/hip_runtime.h>
#include <math.h>

typedef __attribute__((ext_vector_type(16))) __bf16 v16bf;
typedef __attribute__((ext_vector_type(8)))  float  v8f;

union BF16x16 { v16bf v; uint4 q[2]; };

#define NCLUST 64
#define CDIM   512
#define NROWS  1024
#define SPLITS 8
#define ROWS_PER_BLOCK 128
#define CHUNK  32

// ---------------------------------------------------------------------------
// Kernel 1: fused logits-GEMM -> softmax -> aggregation-GEMM, bf16 WMMA.
// grid = 64 batches * 8 splits, block = 256 threads (8 waves, wave32).
// ---------------------------------------------------------------------------
__global__ __launch_bounds__(256) void vlad_gemm_kernel(
    const float* __restrict__ x, const float* __restrict__ centers,
    float* __restrict__ agg_out, float* __restrict__ asum_out)
{
    __shared__ __bf16 s_cent[NCLUST * CDIM];   // 64 KB: centers bf16, [k][c]
    __shared__ __bf16 s_xA[CHUNK * CDIM];      // 32 KB: x chunk row-major [n][c]
    __shared__ __bf16 s_xT[CDIM * CHUNK];      // 32 KB: x chunk transposed [c][n]
    __shared__ float  s_logits[CHUNK * NCLUST];//  8 KB
    __shared__ __bf16 s_aT[NCLUST * CHUNK];    //  4 KB: softmax, transposed [k][n]
    __shared__ float  s_asum[NCLUST];

    const int tid  = threadIdx.x;
    const int wave = tid >> 5;
    const int lane = tid & 31;
    const int lh   = lane >> 4;    // lane half (0/1)
    const int l15  = lane & 15;
    const int b     = blockIdx.x >> 3;
    const int split = blockIdx.x & (SPLITS - 1);

    // Load centers fp32 -> bf16 into LDS (coalesced float4).
    {
        const float4* src = (const float4*)centers;
        for (int i = tid; i < NCLUST * CDIM / 4; i += 256) {
            float4 v = src[i];
            int e = i * 4;
            s_cent[e + 0] = (__bf16)v.x; s_cent[e + 1] = (__bf16)v.y;
            s_cent[e + 2] = (__bf16)v.z; s_cent[e + 3] = (__bf16)v.w;
        }
    }
    if (tid < NCLUST) s_asum[tid] = 0.0f;

    // Phase-B accumulators: 4 cluster-tiles x 4 C-tiles, persist across chunks.
    v8f accB[16];
    #pragma unroll
    for (int t = 0; t < 16; ++t)
        #pragma unroll
        for (int r = 0; r < 8; ++r) accB[t][r] = 0.0f;

    const int rt = wave >> 2;   // phase-A row tile (0..1)
    const int nt = wave & 3;    // phase-A cluster tile (0..3)

    for (int chunk = 0; chunk < ROWS_PER_BLOCK / CHUNK; ++chunk) {
        __syncthreads();  // protect s_xA/s_xT/s_aT from previous iteration readers

        // ---- Cooperative load of 32-row x chunk: fp32 -> bf16, two layouts.
        // Stage 8 float4 global loads in registers per batch so 8
        // global_load_b128 are in flight before the first s_wait_loadcnt.
        const int row0 = split * ROWS_PER_BLOCK + chunk * CHUNK;
        const float4* xsrc = (const float4*)(x + ((size_t)b * NROWS + row0) * CDIM);
        #pragma unroll
        for (int half = 0; half < 2; ++half) {
            float4 buf[8];
            #pragma unroll
            for (int i = 0; i < 8; ++i)
                buf[i] = xsrc[tid + (half * 8 + i) * 256];
            #pragma unroll
            for (int i = 0; i < 8; ++i) {
                int e  = (tid + (half * 8 + i) * 256) * 4;
                int rr = e >> 9;       // local row (CDIM=512 elems/row)
                int cc = e & 511;
                s_xA[e + 0] = (__bf16)buf[i].x; s_xA[e + 1] = (__bf16)buf[i].y;
                s_xA[e + 2] = (__bf16)buf[i].z; s_xA[e + 3] = (__bf16)buf[i].w;
                s_xT[(cc + 0) * CHUNK + rr] = (__bf16)buf[i].x;
                s_xT[(cc + 1) * CHUNK + rr] = (__bf16)buf[i].y;
                s_xT[(cc + 2) * CHUNK + rr] = (__bf16)buf[i].z;
                s_xT[(cc + 3) * CHUNK + rr] = (__bf16)buf[i].w;
            }
        }
        // Fire-and-forget L2 prefetch of the next chunk; lands during the
        // WMMA/softmax phases below (emits global_prefetch_b8, no counter).
        if (chunk + 1 < ROWS_PER_BLOCK / CHUNK) {
            const char* nxt = (const char*)(xsrc + CHUNK * CDIM / 4);
            #pragma unroll
            for (int i = 0; i < 16; ++i)
                __builtin_prefetch(nxt + ((size_t)(tid + i * 256)) * 16, 0, 0);
        }
        __syncthreads();

        // ---- Phase A: logits tile (16 rows x 16 clusters) per wave ----
        {
            v8f acc = {};
            #pragma unroll
            for (int step = 0; step < CDIM / 32; ++step) {
                BF16x16 A, Bm;
                // A (16x32 bf16): lane l15 = row; e0..7 -> k=lh*8.., e8..15 -> +16
                const __bf16* pa = s_xA + (rt * 16 + l15) * CDIM + step * 32 + lh * 8;
                A.q[0] = *(const uint4*)(pa);
                A.q[1] = *(const uint4*)(pa + 16);
                // B (32x16 bf16) = centers^T: lane l15 = cluster col; e = c offset
                const __bf16* pb = s_cent + (nt * 16 + l15) * CDIM + step * 32 + lh * 8;
                Bm.q[0] = *(const uint4*)(pb);
                Bm.q[1] = *(const uint4*)(pb + 16);
                acc = __builtin_amdgcn_wmma_f32_16x16x32_bf16(
                        false, A.v, false, Bm.v, (short)0, acc, false, false);
            }
            #pragma unroll
            for (int r = 0; r < 8; ++r)
                s_logits[(rt * 16 + r + lh * 8) * NCLUST + nt * 16 + l15] = acc[r];
        }
        __syncthreads();

        // ---- Softmax over 64 clusters, one thread per local row ----
        if (tid < CHUNK) {
            const int n = tid;
            float m = -3.0e38f;
            for (int k = 0; k < NCLUST; ++k) m = fmaxf(m, s_logits[n * NCLUST + k]);
            float s = 0.0f;
            for (int k = 0; k < NCLUST; ++k) s += __expf(s_logits[n * NCLUST + k] - m);
            float inv = 1.0f / s;
            for (int k = 0; k < NCLUST; ++k) {
                float a = __expf(s_logits[n * NCLUST + k] - m) * inv;
                s_aT[k * CHUNK + n] = (__bf16)a;
                atomicAdd(&s_asum[k], a);
            }
        }
        __syncthreads();

        // ---- Phase B: agg += a^T (64 x 32) * x (32 x 512), one K-step ----
        {
            BF16x16 Aa[4];
            #pragma unroll
            for (int mt = 0; mt < 4; ++mt) {
                const __bf16* pa = s_aT + (mt * 16 + l15) * CHUNK + lh * 8;
                Aa[mt].q[0] = *(const uint4*)(pa);
                Aa[mt].q[1] = *(const uint4*)(pa + 16);
            }
            #pragma unroll
            for (int ct = 0; ct < 4; ++ct) {
                const int c0 = wave * 64 + ct * 16;
                BF16x16 Bx;
                const __bf16* pb = s_xT + (c0 + l15) * CHUNK + lh * 8;
                Bx.q[0] = *(const uint4*)(pb);
                Bx.q[1] = *(const uint4*)(pb + 16);
                #pragma unroll
                for (int mt = 0; mt < 4; ++mt)
                    accB[mt * 4 + ct] = __builtin_amdgcn_wmma_f32_16x16x32_bf16(
                        false, Aa[mt].v, false, Bx.v, (short)0, accB[mt * 4 + ct],
                        false, false);
            }
        }
    }

    // Flush partial agg (C/D layout: VGPR r -> m = r + lh*8, lane l15 -> col)
    #pragma unroll
    for (int mt = 0; mt < 4; ++mt)
        #pragma unroll
        for (int ct = 0; ct < 4; ++ct)
            #pragma unroll
            for (int r = 0; r < 8; ++r) {
                int k = mt * 16 + r + lh * 8;
                int c = wave * 64 + ct * 16 + l15;
                unsafeAtomicAdd(&agg_out[((size_t)b * NCLUST + k) * CDIM + c],
                                accB[mt * 4 + ct][r]);
            }
    __syncthreads();
    if (tid < NCLUST) unsafeAtomicAdd(&asum_out[b * NCLUST + tid], s_asum[tid]);
}

// ---------------------------------------------------------------------------
// Kernel 2: vlad = agg - a_sum*centers, then global L2-normalize per batch.
// ---------------------------------------------------------------------------
__global__ __launch_bounds__(256) void vlad_finalize_kernel(
    float* __restrict__ out, const float* __restrict__ centers,
    const float* __restrict__ asum)
{
    __shared__ float s_asum[NCLUST];
    __shared__ float s_red[8];
    __shared__ float s_inv;
    const int tid = threadIdx.x;
    const int b   = blockIdx.x;
    if (tid < NCLUST) s_asum[tid] = asum[b * NCLUST + tid];
    __syncthreads();

    float* po = out + (size_t)b * (NCLUST * CDIM);
    float sq = 0.0f;
    for (int i = tid; i < NCLUST * CDIM; i += 256) {
        int k = i >> 9;
        int c = i & 511;
        float v = po[i] - s_asum[k] * centers[k * CDIM + c];
        po[i] = v;
        sq += v * v;
    }
    #pragma unroll
    for (int off = 16; off > 0; off >>= 1) sq += __shfl_down(sq, off, 32);
    if ((tid & 31) == 0) s_red[tid >> 5] = sq;
    __syncthreads();
    if (tid == 0) {
        float t = 0.0f;
        for (int w = 0; w < 8; ++w) t += s_red[w];
        s_inv = rsqrtf(fmaxf(t, 1e-12f));
    }
    __syncthreads();
    float inv = s_inv;
    for (int i = tid; i < NCLUST * CDIM; i += 256) po[i] *= inv;
}

// ---------------------------------------------------------------------------
extern "C" void kernel_launch(void* const* d_in, const int* in_sizes, int n_in,
                              void* d_out, int out_size, void* d_ws, size_t ws_size,
                              hipStream_t stream) {
    const float* x       = (const float*)d_in[0];   // (64,16,64,512) f32
    const float* centers = (const float*)d_in[1];   // (64,512) f32
    float* out  = (float*)d_out;                    // (64, 64*512) f32
    float* asum = (float*)d_ws;                     // (64,64) f32 scratch

    hipMemsetAsync(d_out, 0, sizeof(float) * (size_t)out_size, stream);
    hipMemsetAsync(d_ws, 0, sizeof(float) * 64 * NCLUST, stream);

    vlad_gemm_kernel<<<dim3(64 * SPLITS), dim3(256), 0, stream>>>(x, centers, out, asum);
    vlad_finalize_kernel<<<dim3(64), dim3(256), 0, stream>>>(out, centers, asum);
}
